// regression_model_5823975653896
// MI455X (gfx1250) — matrix operand order repaired and checked by reading, not compiled, not run on hardware
//
#include <hip/hip_runtime.h>
#include <hip/hip_bf16.h>
#include <math.h>

typedef _Float16 half_t;
typedef __attribute__((ext_vector_type(16))) _Float16 v16h;
typedef __attribute__((ext_vector_type(8)))  float    v8f;
typedef __attribute__((ext_vector_type(4)))  unsigned int v4u;
typedef __attribute__((ext_vector_type(8)))  int      v8i;
typedef __attribute__((ext_vector_type(4)))  int      v4i;

// ---- TDM availability (device pass only; host pass uses fallback path) ----
#if defined(__HIP_DEVICE_COMPILE__) && defined(__has_builtin)
#if __has_builtin(__builtin_amdgcn_tensor_load_to_lds) && __has_builtin(__builtin_amdgcn_s_wait_tensorcnt)
#define HAVE_TDM 1
#endif
#endif
#ifndef HAVE_TDM
#define HAVE_TDM 0
#endif

// ---------------- WMMA helpers (CDNA5 v_wmma_f32_16x16x32_f16) ----------------

static __device__ __forceinline__ v8f wmma_f32_f16(v16h a, v16h b, v8f c) {
  return __builtin_amdgcn_wmma_f32_16x16x32_f16(false, a, false, b, (short)0, c, false, false);
}

// A matrix 16x32 f16, row-major source [M][lda].
// Lane L holds row M = m0 + (L%16); half i holds K = k0 + (L/16)*16 + i.
template <typename P>
static __device__ __forceinline__ v16h load_a16(P A, int lda, int m0, int k0, int lane) {
  int row = m0 + (lane & 15);
  int hi  = (lane >> 4) & 1;
  P p = A + (size_t)row * lda + k0 + hi * 16;
  v16h a;
#pragma unroll
  for (int i = 0; i < 16; ++i) a[i] = p[i];
  return a;
}

// B matrix 32x16 f16 from memory stored [n][k] (k contiguous, pitch ldb).
// Lane L holds col N = n0 + (L%16); half i holds K = k0 + (L/16)*16 + i.
template <typename P>
static __device__ __forceinline__ v16h load_b16(P Bm, int ldb, int n0, int k0, int lane) {
  P p = Bm + (size_t)(n0 + (lane & 15)) * ldb + k0 + ((lane >> 4) & 1) * 16;
  v16h b;
#pragma unroll
  for (int i = 0; i < 16; ++i) b[i] = p[i];
  return b;
}

#if HAVE_TDM
// 2D tile Global->LDS via Tensor Data Mover.
// D# group0: count=1, lds_addr, 57b global_addr, type=2. group1: data_size=2B,
// tensor_dim0 (=remaining elems along contiguous axis; OOB reads return 0),
// tensor_dim1, tile_dim0/1, tensor_dim0_stride (elements).
static __device__ __forceinline__ void tdm_load_2d(unsigned lds_off, const void* gptr,
                                                   unsigned dim0, unsigned dim1,
                                                   unsigned tile0, unsigned tile1,
                                                   unsigned long long stride0) {
  unsigned long long ga = (unsigned long long)gptr;
  v4u g0;
  g0[0] = 1u;                                                   // count=1 (valid)
  g0[1] = lds_off;                                              // lds_addr
  g0[2] = (unsigned)(ga & 0xffffffffu);                         // global_addr[31:0]
  g0[3] = (unsigned)((ga >> 32) & 0x1ffffffu) | (2u << 30);     // addr[56:32] | type=2
  v8i g1;
  g1[0] = (int)(1u << 16);                                      // data_size=1 (2 bytes)
  g1[1] = (int)((dim0 & 0xffffu) << 16);                        // tensor_dim0[15:0] @63:48
  g1[2] = (int)(((dim0 >> 16) & 0xffffu) | ((dim1 & 0xffffu) << 16));
  g1[3] = (int)(((dim1 >> 16) & 0xffffu) | ((tile0 & 0xffffu) << 16)); // tile_dim0 @127:112
  g1[4] = (int)(tile1 & 0xffffu);                               // tile_dim1; tile_dim2=0
  g1[5] = (int)(unsigned)(stride0 & 0xffffffffull);             // dim0_stride[31:0]
  g1[6] = (int)(unsigned)((stride0 >> 32) & 0xffffull);         // dim0_stride[47:32]
  g1[7] = 0;
  v4i z4 = {0, 0, 0, 0};
#if __clang_major__ >= 23
  v8i z8 = {0, 0, 0, 0, 0, 0, 0, 0};
  __builtin_amdgcn_tensor_load_to_lds(g0, g1, z4, z4, z8, 0);
#else
  __builtin_amdgcn_tensor_load_to_lds(g0, g1, z4, z4, 0);
#endif
}
#endif

static __device__ __forceinline__ float sigmoidf_(float x) { return 1.f / (1.f + expf(-x)); }

static __device__ __forceinline__ int frame_num_dev(int l) {
  int f = (l - 250) / 3; f /= 4; f /= 5; f /= 2; f /= 2; return f;
}

// ---------------- small elementwise / prep kernels ----------------

__global__ void k_cvt_f16(const float* __restrict__ s, half_t* __restrict__ d, size_t n) {
  size_t st = (size_t)gridDim.x * blockDim.x;
  for (size_t i = (size_t)blockIdx.x * blockDim.x + threadIdx.x; i < n; i += st)
    d[i] = (half_t)s[i];
}

__global__ void k_zero_f32(float* __restrict__ d, size_t n) {
  size_t st = (size_t)gridDim.x * blockDim.x;
  for (size_t i = (size_t)blockIdx.x * blockDim.x + threadIdx.x; i < n; i += st)
    d[i] = 0.f;
}

__global__ void k_relu_cvt(const float* __restrict__ s, half_t* __restrict__ d, size_t n) {
  size_t st = (size_t)gridDim.x * blockDim.x;
  for (size_t i = (size_t)blockIdx.x * blockDim.x + threadIdx.x; i < n; i += st)
    d[i] = (half_t)fmaxf(s[i], 0.f);
}

// coatt_W [d][e] f32 -> transposed f16 [e][d]
__global__ void k_cvt_T256(const float* __restrict__ s, half_t* __restrict__ d) {
  size_t st = (size_t)gridDim.x * blockDim.x;
  for (size_t i = (size_t)blockIdx.x * blockDim.x + threadIdx.x; i < 256 * 256; i += st) {
    int e = (int)(i >> 8), dd = (int)(i & 255);
    d[i] = (half_t)s[(size_t)dd * 256 + e];
  }
}

// sinc weights [64][251] f32 -> [64][256] f16 zero-padded (K multiple of 32)
__global__ void k_cvt_sinc(const float* __restrict__ s, half_t* __restrict__ d) {
  size_t st = (size_t)gridDim.x * blockDim.x;
  for (size_t i = (size_t)blockIdx.x * blockDim.x + threadIdx.x; i < 64 * 256; i += st) {
    int o = (int)(i >> 8), j = (int)(i & 255);
    d[i] = (j < 251) ? (half_t)s[(size_t)o * 251 + j] : (half_t)0.f;
  }
}

// pack LSTM weights: W[g][k] = k<64 ? Wih[g][k] : Whh[g][k-64]  (f16), bias = bih+bhh (f32)
__global__ void k_lstm_pack(const float* __restrict__ Wih, const float* __restrict__ Whh,
                            const float* __restrict__ bih, const float* __restrict__ bhh,
                            half_t* __restrict__ W, float* __restrict__ bb) {
  size_t st = (size_t)gridDim.x * blockDim.x;
  size_t t0 = (size_t)blockIdx.x * blockDim.x + threadIdx.x;
  for (size_t i = t0; i < (size_t)512 * 192; i += st) {
    int g = (int)(i / 192), k = (int)(i % 192);
    W[i] = (half_t)(k < 64 ? Wih[(size_t)g * 64 + k] : Whh[(size_t)g * 128 + (k - 64)]);
  }
  for (size_t g = t0; g < 512; g += st) bb[g] = bih[g] + bhh[g];
}

// ---------------- SincNet front-end as WMMA GEMM (K=251 padded to 256) ----------------

__global__ __launch_bounds__(256) void k_sinc_wmma(
    const float* __restrict__ x1, const float* __restrict__ x2,
    const half_t* __restrict__ Wh,      // [64][256] f16 padded
    const float* __restrict__ bias,
    half_t* __restrict__ outh, int T) {
  __shared__ half_t im[64][264];        // [t][j]
  int b = blockIdx.y;
  int t0 = blockIdx.x * 64;
  int tid = threadIdx.x;
  const float* xs = (b < 4) ? (x1 + (size_t)b * 48000) : (x2 + (size_t)(b - 4) * 48000);
  for (int e = tid; e < 64 * 256; e += 256) {
    int t = e & 63, j = e >> 6;
    int tg = t0 + t;
    im[t][j] = (tg < T && j < 251) ? (half_t)xs[3 * tg + j] : (half_t)0.f;
  }
  __syncthreads();
  int lane = tid & 31, w = tid >> 5;
  int m = w & 3;
  int hi = (lane >> 4) & 1, col = lane & 15;
#pragma unroll
  for (int nn = 0; nn < 2; ++nn) {
    int n = (w >> 2) + nn * 2;
    v8f acc;
#pragma unroll
    for (int r = 0; r < 8; ++r) acc[r] = bias[m * 16 + r + 8 * hi];
#pragma unroll
    for (int ks = 0; ks < 8; ++ks)
      acc = wmma_f32_f16(load_a16(Wh, 256, m * 16, ks * 32, lane),
                         load_b16(&im[0][0], 264, n * 16, ks * 32, lane), acc);
    int t = t0 + n * 16 + col;
    if (t < T) {
#pragma unroll
      for (int r = 0; r < 8; ++r) {
        float v = acc[r];
        if (v < 0.f) v *= 0.2f;                 // leaky relu 0.2
        outh[((size_t)b * 64 + (m * 16 + r + 8 * hi)) * T + t] = (half_t)v;
      }
    }
  }
}

// ---------------- 1x1 conv: Y[64][T] = W[64][64] @ X[64][T] + b (TDM + WMMA) ----------------

__global__ __launch_bounds__(256) void k_conv1x1(
    const half_t* __restrict__ Xh, const half_t* __restrict__ Wh,
    const float* __restrict__ bias, float* __restrict__ Yf, half_t* __restrict__ Yh,
    int T, int relu) {
  __shared__ half_t smem[64 * 64 + 64 * 72];   // xT tile [c][t] (TDM dest, offset 0) + xB [t][c]
  half_t* xT = smem;
  half_t (*xB)[72] = (half_t(*)[72])(smem + 64 * 64);
  int b = blockIdx.y;
  int t0 = blockIdx.x * 64;
  int tid = threadIdx.x;
  const half_t* Xb = Xh + (size_t)b * 64 * T;
#if HAVE_TDM
  // DMA the [64c][64t] tile into LDS: contiguous axis = t, stride T, OOB -> zero fill.
  if (tid < 32) {                               // wave 0 issues the tensor DMA
    tdm_load_2d(/*lds_off=*/0u, Xb + t0, (unsigned)(T - t0), 64u, 64u, 64u,
                (unsigned long long)T);
    __builtin_amdgcn_s_wait_tensorcnt(0);
  }
  __syncthreads();
  for (int e = tid; e < 64 * 64; e += 256) {    // LDS transpose -> WMMA B layout [t][c]
    int t = e & 63, c = e >> 6;
    xB[t][c] = xT[(size_t)c * 64 + t];
  }
  __syncthreads();
#else
  {
    int c = tid >> 2;
    int tq = (tid & 3) * 16;
    const half_t* row = Xb + (size_t)c * T;
#pragma unroll
    for (int i = 0; i < 16; ++i) {
      int t = t0 + tq + i;
      xB[tq + i][c] = (t < T) ? row[t] : (half_t)0.f;
    }
  }
  __syncthreads();
#endif
  int lane = tid & 31, w = tid >> 5;
  int m = w & 3;
  int hi = (lane >> 4) & 1, col = lane & 15;
  v16h a0 = load_a16(Wh, 64, m * 16, 0, lane);
  v16h a1 = load_a16(Wh, 64, m * 16, 32, lane);
#pragma unroll
  for (int nn = 0; nn < 2; ++nn) {
    int n = (w >> 2) + nn * 2;
    v8f acc;
#pragma unroll
    for (int r = 0; r < 8; ++r) acc[r] = bias[m * 16 + r + 8 * hi];
    acc = wmma_f32_f16(a0, load_b16(&xB[0][0], 72, n * 16, 0, lane), acc);
    acc = wmma_f32_f16(a1, load_b16(&xB[0][0], 72, n * 16, 32, lane), acc);
    int t = t0 + n * 16 + col;
    if (t < T) {
#pragma unroll
      for (int r = 0; r < 8; ++r) {
        float v = acc[r];
        if (relu) v = fmaxf(v, 0.f);
        size_t oi = ((size_t)b * 64 + (m * 16 + r + 8 * hi)) * T + t;
        if (Yf) Yf[oi] = v;
        if (Yh) Yh[oi] = (half_t)v;
      }
    }
  }
}

// ---------------- fused WaveNet block ----------------
// gate = Wg[128][192] @ im2col(x)  ->  out = tanh*sig  ->  s += Wsk@out+bsk ; x' = (Wrs@out+brs + x)*0.707

__global__ __launch_bounds__(256) void k_wn_block(
    const half_t* __restrict__ Xh, const float* __restrict__ Xf,
    const half_t* __restrict__ Wg, const float* __restrict__ bg,
    const half_t* __restrict__ Wsk, const float* __restrict__ bsk,
    const half_t* __restrict__ Wrs, const float* __restrict__ brs,
    float* __restrict__ S, float* __restrict__ Xof, half_t* __restrict__ Xoh,
    int T, int dil) {
  __shared__ half_t im[64][200];   // im2col [t][k], k = i*3 + tap
  __shared__ half_t gt[128][72];   // gate   [o][t]
  __shared__ half_t ob[64][72];    // out    [t][c]  (B layout for skip/res)
  int b = blockIdx.y;
  int t0 = blockIdx.x * 64;
  int tid = threadIdx.x;
  const half_t* Xb = Xh + (size_t)b * 64 * T;
  for (int e = tid; e < 64 * 192; e += 256) {
    int t = e & 63, k = e >> 6;
    int i = k / 3, k3 = k - 3 * i;
    int tg = t0 + t + (k3 - 1) * dil;
    bool ok = (tg >= 0 && tg < T);
    im[t][k] = ok ? Xb[(size_t)i * T + tg] : (half_t)0.f;
    if (ok && t == 0) __builtin_prefetch(Xb + (size_t)i * T + tg + 64, 0, 1); // next tile
  }
  __syncthreads();
  int lane = tid & 31, w = tid >> 5;
  int hi = (lane >> 4) & 1, col = lane & 15;
  { // gate GEMM: wave w handles M-tile m=w, N-tiles 0..3, K=192
    int m = w;
    v16h av[6];
#pragma unroll
    for (int ks = 0; ks < 6; ++ks) av[ks] = load_a16(Wg, 192, m * 16, ks * 32, lane);
#pragma unroll
    for (int n = 0; n < 4; ++n) {
      v8f acc;
#pragma unroll
      for (int r = 0; r < 8; ++r) acc[r] = bg[m * 16 + r + 8 * hi];
#pragma unroll
      for (int ks = 0; ks < 6; ++ks)
        acc = wmma_f32_f16(av[ks], load_b16(&im[0][0], 200, n * 16, ks * 32, lane), acc);
#pragma unroll
      for (int r = 0; r < 8; ++r) gt[m * 16 + r + 8 * hi][n * 16 + col] = (half_t)acc[r];
    }
  }
  __syncthreads();
  for (int e = tid; e < 64 * 64; e += 256) {   // gating
    int t = e & 63, c = e >> 6;
    float xt = (float)gt[c][t];
    float xs = (float)gt[c + 64][t];
    ob[t][c] = (half_t)(tanhf(xt) * sigmoidf_(xs));
  }
  __syncthreads();
#pragma unroll
  for (int p = 0; p < 2; ++p) {  // skip + res GEMMs: 16 tiles over 8 waves
    int tI = w + p * 8;
    int m = tI >> 2, n = tI & 3;
    int t = t0 + n * 16 + col;
    v16h b0 = load_b16(&ob[0][0], 72, n * 16, 0, lane);
    v16h b1 = load_b16(&ob[0][0], 72, n * 16, 32, lane);
    { // skip
      v8f acc;
#pragma unroll
      for (int r = 0; r < 8; ++r) acc[r] = bsk[m * 16 + r + 8 * hi];
      acc = wmma_f32_f16(load_a16(Wsk, 64, m * 16, 0, lane), b0, acc);
      acc = wmma_f32_f16(load_a16(Wsk, 64, m * 16, 32, lane), b1, acc);
      if (t < T) {
#pragma unroll
        for (int r = 0; r < 8; ++r) {
          size_t oi = ((size_t)b * 64 + (m * 16 + r + 8 * hi)) * T + t;
          S[oi] += acc[r];
        }
      }
    }
    { // res + residual scale
      v8f acc;
#pragma unroll
      for (int r = 0; r < 8; ++r) acc[r] = brs[m * 16 + r + 8 * hi];
      acc = wmma_f32_f16(load_a16(Wrs, 64, m * 16, 0, lane), b0, acc);
      acc = wmma_f32_f16(load_a16(Wrs, 64, m * 16, 32, lane), b1, acc);
      if (t < T) {
#pragma unroll
        for (int r = 0; r < 8; ++r) {
          size_t oi = ((size_t)b * 64 + (m * 16 + r + 8 * hi)) * T + t;
          float xn = (acc[r] + Xf[oi]) * 0.707f;
          Xof[oi] = xn;
          Xoh[oi] = (half_t)xn;
        }
      }
    }
  }
}

// ---------------- max pool ----------------

__global__ void k_pool(const float* __restrict__ X, float* __restrict__ Yf,
                       half_t* __restrict__ Yh, int Tin, int k, int Tout) {
  size_t n = (size_t)8 * 64 * Tout;
  size_t st = (size_t)gridDim.x * blockDim.x;
  for (size_t idx = (size_t)blockIdx.x * blockDim.x + threadIdx.x; idx < n; idx += st) {
    int t = (int)(idx % Tout);
    size_t bc = idx / Tout;
    const float* p = X + bc * Tin + (size_t)t * k;
    float m = p[0];
    for (int j = 1; j < k; ++j) m = fmaxf(m, p[j]);
    Yf[idx] = m;
    Yh[idx] = (half_t)m;
  }
}

// ---------------- bidirectional LSTM (persistent, WMMA gates) ----------------

__global__ __launch_bounds__(256) void k_lstm(
    const half_t* __restrict__ Yin,   // [8][64][T]
    const half_t* __restrict__ Wp,    // [2][512][192]
    const float* __restrict__ bp,     // [2][512]
    float* __restrict__ Enc,          // [8][T][256]
    int T) {
  __shared__ half_t zB[16][200];      // [n][k]; rows 8..15 zero
  __shared__ float  gl[512][9];       // gates [g][b]
  int dir = blockIdx.x;
  const half_t* W = Wp + (size_t)dir * 512 * 192;
  const float* bias = bp + (size_t)dir * 512;
  int tid = threadIdx.x, lane = tid & 31, w = tid >> 5;
  int hi = (lane >> 4) & 1, col = lane & 15;
  float cst[4];
#pragma unroll
  for (int q = 0; q < 4; ++q) cst[q] = 0.f;
  for (int e = tid; e < 16 * 200; e += 256) (&zB[0][0])[e] = (half_t)0.f;
  __syncthreads();
  for (int it = 0; it < T; ++it) {
    int t = dir ? (T - 1 - it) : it;
    for (int e = tid; e < 512; e += 256) {       // x_t part
      int n = e >> 6, c = e & 63;
      zB[n][c] = Yin[((size_t)n * 64 + c) * T + t];
    }
    __syncthreads();
#pragma unroll
    for (int mi = 0; mi < 4; ++mi) {             // 32 M-tiles over 8 waves
      int m = w + mi * 8;
      v8f acc;
#pragma unroll
      for (int r = 0; r < 8; ++r) acc[r] = bias[m * 16 + r + 8 * hi];
#pragma unroll
      for (int ks = 0; ks < 6; ++ks)
        acc = wmma_f32_f16(load_a16(W, 192, m * 16, ks * 32, lane),
                           load_b16(&zB[0][0], 200, 0, ks * 32, lane), acc);
      if (col < 8) {
#pragma unroll
        for (int r = 0; r < 8; ++r) gl[m * 16 + r + 8 * hi][col] = acc[r];
      }
    }
    __syncthreads();
#pragma unroll
    for (int q = 0; q < 4; ++q) {                // gating: i,f,g,o
      int e = tid + q * 256;
      int bb = e >> 7, h = e & 127;
      float ii = gl[h][bb], ff = gl[128 + h][bb], gg = gl[256 + h][bb], oo = gl[384 + h][bb];
      float c = sigmoidf_(ff) * cst[q] + sigmoidf_(ii) * tanhf(gg);
      cst[q] = c;
      float hv = sigmoidf_(oo) * tanhf(c);
      Enc[((size_t)bb * T + t) * 256 + dir * 128 + h] = hv;
      zB[bb][64 + h] = (half_t)hv;
    }
    __syncthreads();
  }
}

// ---------------- enc padding / transposes for attention GEMMs ----------------

__global__ void k_encpack(const float* __restrict__ Enc, half_t* __restrict__ EncH,
                          half_t* __restrict__ EncTH) {
  size_t n = (size_t)8 * 224 * 256;
  size_t st = (size_t)gridDim.x * blockDim.x;
  for (size_t idx = (size_t)blockIdx.x * blockDim.x + threadIdx.x; idx < n; idx += st) {
    int e = (int)(idx & 255);
    int r = (int)(idx >> 8);
    int t = r % 224, b = r / 224;
    float v = (t < 198) ? Enc[((size_t)b * 198 + t) * 256 + e] : 0.f;
    EncH[((size_t)b * 224 + t) * 256 + e] = (half_t)v;
    EncTH[((size_t)b * 256 + e) * 224 + t] = (half_t)v;
  }
}

// ---------------- generic batched WMMA GEMM: one 16x16 tile per wave-block ----------------

__global__ __launch_bounds__(32) void k_gemm(
    const half_t* __restrict__ A, int lda, long long sA,
    const half_t* __restrict__ Bm, int ldb, long long sB,
    float* __restrict__ Cf, half_t* __restrict__ Ch, int ldc, long long sC, int K) {
  int lane = threadIdx.x;
  long long nb = blockIdx.z;
  const half_t* Ab = A + nb * sA;
  const half_t* Bb = Bm + nb * sB;
  int m0 = blockIdx.y * 16, n0 = blockIdx.x * 16;
  v8f acc;
#pragma unroll
  for (int r = 0; r < 8; ++r) acc[r] = 0.f;
  for (int k0 = 0; k0 < K; k0 += 32)
    acc = wmma_f32_f16(load_a16(Ab, lda, m0, k0, lane), load_b16(Bb, ldb, n0, k0, lane), acc);
  int hi = (lane >> 4) & 1, col = lane & 15;
#pragma unroll
  for (int r = 0; r < 8; ++r) {
    long long oi = nb * sC + (long long)(m0 + r + 8 * hi) * ldc + n0 + col;
    if (Cf) Cf[oi] = acc[r];
    if (Ch) Ch[oi] = (half_t)acc[r];
  }
}

// ---------------- masked softmaxes ----------------

__global__ __launch_bounds__(256) void k_softmax2(const float* __restrict__ Sim,
                                                  half_t* __restrict__ P,
                                                  const int* __restrict__ len1,
                                                  const int* __restrict__ len2) {
  __shared__ float red[256];
  int b = blockIdx.y, t = blockIdx.x, tid = threadIdx.x;
  int f1 = frame_num_dev(len1[b]), f2 = frame_num_dev(len2[b]);
  const float* row = Sim + ((size_t)b * 224 + t) * 224;
  float v = -1e9f;
  if (tid < 224) v = (t < f1 && tid < f2) ? row[tid] : -1e9f;
  red[tid] = v; __syncthreads();
  for (int s2 = 128; s2 > 0; s2 >>= 1) { if (tid < s2) red[tid] = fmaxf(red[tid], red[tid + s2]); __syncthreads(); }
  float mx = red[0]; __syncthreads();
  float ex = (tid < 224) ? expf(v - mx) : 0.f;
  red[tid] = ex; __syncthreads();
  for (int s2 = 128; s2 > 0; s2 >>= 1) { if (tid < s2) red[tid] += red[tid + s2]; __syncthreads(); }
  float sm = red[0]; __syncthreads();
  if (tid < 224) P[((size_t)b * 224 + t) * 224 + tid] = (half_t)(ex / sm);
}

__global__ __launch_bounds__(256) void k_softmax1T(const float* __restrict__ Sim,
                                                   half_t* __restrict__ P,
                                                   const int* __restrict__ len1,
                                                   const int* __restrict__ len2) {
  __shared__ float red[256];
  int b = blockIdx.y, s = blockIdx.x, tid = threadIdx.x;
  int f1 = frame_num_dev(len1[b]), f2 = frame_num_dev(len2[b]);
  float v = -1e9f;
  if (tid < 224) v = (tid < f1 && s < f2) ? Sim[((size_t)b * 224 + tid) * 224 + s] : -1e9f;
  red[tid] = v; __syncthreads();
  for (int s2 = 128; s2 > 0; s2 >>= 1) { if (tid < s2) red[tid] = fmaxf(red[tid], red[tid + s2]); __syncthreads(); }
  float mx = red[0]; __syncthreads();
  float ex = (tid < 224) ? expf(v - mx) : 0.f;
  red[tid] = ex; __syncthreads();
  for (int s2 = 128; s2 > 0; s2 >>= 1) { if (tid < s2) red[tid] += red[tid + s2]; __syncthreads(); }
  float sm = red[0]; __syncthreads();
  if (tid < 224) P[((size_t)b * 224 + s) * 224 + tid] = (half_t)(ex / sm);
}

// ---------------- masked sums -> d1/d2 ----------------

__global__ void k_dsum(const float* __restrict__ A1, const float* __restrict__ A2,
                       const float* __restrict__ Enc, const int* __restrict__ len1,
                       const int* __restrict__ len2, float* __restrict__ DD) {
  int idx = blockIdx.x * 256 + threadIdx.x;
  if (idx >= 2048) return;
  int which = idx >> 10;
  int rem = idx & 1023;
  int b = rem >> 8, d = rem & 255;
  int fn = frame_num_dev(which ? len2[b] : len1[b]);
  const float* A = (which ? A2 : A1) + (size_t)b * 224 * 256 + d;
  float sa = 0.f;
  for (int t = 0; t < fn; ++t) sa += A[(size_t)t * 256];
  const float* E = Enc + (size_t)(which ? b + 4 : b) * 198 * 256 + d;
  float se = 0.f;
  for (int t = 0; t < 198; ++t) se += E[(size_t)t * 256];
  DD[idx] = fabsf(sa - se) / (float)fn;
}

// ---------------- MLP head + loss ----------------

__global__ __launch_bounds__(128) void k_head(const float* __restrict__ DD,
                                              const float* __restrict__ W1,
                                              const float* __restrict__ b1,
                                              const float* __restrict__ W2,
                                              const float* __restrict__ b2,
                                              const float* __restrict__ label,
                                              float* __restrict__ out) {
  __shared__ float red[128];
  int tid = threadIdx.x;
  float loss = 0.f;
  for (int b = 0; b < 4; ++b) {
    float sb = 0.f;
    for (int which = 0; which < 2; ++which) {
      const float* dv = DD + which * 1024 + b * 256;
      float acc = b1[tid];
      for (int d = 0; d < 256; ++d) acc += W1[(size_t)tid * 256 + d] * dv[d];
      acc = fmaxf(acc, 0.f);
      red[tid] = acc * W2[tid];
      __syncthreads();
      for (int s2 = 64; s2 > 0; s2 >>= 1) { if (tid < s2) red[tid] += red[tid + s2]; __syncthreads(); }
      float r = red[0] + b2[0];
      __syncthreads();
      sb += r;
    }
    sb *= 0.5f;
    if (tid == 0) out[1 + b] = sb;
    float diff = label[b] - sb;
    loss += diff * diff;
  }
  if (tid == 0) out[0] = loss * 0.25f;
}

// ---------------- host ----------------

static inline unsigned gsb(size_t n) {
  size_t b = (n + 255) / 256;
  if (b > 16384) b = 16384;
  return (unsigned)b;
}

extern "C" void kernel_launch(void* const* d_in, const int* in_sizes, int n_in,
                              void* d_out, int out_size, void* d_ws, size_t ws_size,
                              hipStream_t stream) {
  (void)in_sizes; (void)n_in; (void)out_size; (void)ws_size;
  auto f = [&](int i) { return (const float*)d_in[i]; };
  auto li = [&](int i) { return (const int*)d_in[i]; };

  // param indices (jax pytree: sorted dict keys)
  // 0:x1 1:x2 2:lenX1 3:lenX2 4:label 5:W1 6:W2 7:b1 8:b2 9:coatt_W
  // 10:Whh_b 11:Whh_f 12:Wih_b 13:Wih_f 14:bhh_b 15:bhh_f 16:bih_b 17:bih_f
  // 18:sinc_b 19:sinc_w 20..: wavenet[4]{blocks[7]{in_w,in_b,res_w,res_b,skip_w,skip_b},first,last1,last2}
  const int WN0 = 20;

  const int T1 = 15917;
  const int Ts[4] = {15917, 3979, 795, 397};
  const int Kp[4] = {4, 5, 2, 2};
  const int Tp[4] = {3979, 795, 397, 198};
  const size_t NBT1 = (size_t)8 * 64 * T1;

  char* ws = (char*)d_ws;
  size_t off = 0;
  auto alloc = [&](size_t bytes) -> void* {
    off = (off + 255) & ~(size_t)255;
    void* p = (void*)(ws + off);
    off += bytes;
    return p;
  };
  float*  A32   = (float*)alloc(NBT1 * 4);
  float*  B32   = (float*)alloc(NBT1 * 4);
  float*  S32   = (float*)alloc(NBT1 * 4);
  half_t* A16   = (half_t*)alloc(NBT1 * 2);
  half_t* B16   = (half_t*)alloc(NBT1 * 2);
  half_t* PINH  = (half_t*)alloc(NBT1 * 2);
  float*  PIN32 = (float*)alloc((size_t)8 * 64 * 3979 * 4);
  const size_t WG_H = 128 * 192, W11_H = 64 * 64;
  const size_t BLK_H = WG_H + 2 * W11_H;
  const size_t STG_H = 7 * BLK_H + 3 * W11_H;
  half_t* WWN  = (half_t*)alloc(4 * STG_H * 2);
  half_t* WSIN = (half_t*)alloc((size_t)64 * 256 * 2);
  half_t* WL   = (half_t*)alloc((size_t)2 * 512 * 192 * 2);
  float*  BL   = (float*)alloc((size_t)2 * 512 * 4);
  half_t* WCT  = (half_t*)alloc((size_t)256 * 256 * 2);
  float*  ENC  = (float*)alloc((size_t)8 * 198 * 256 * 4);
  half_t* ENCH = (half_t*)alloc((size_t)8 * 224 * 256 * 2);
  half_t* ENCTH= (half_t*)alloc((size_t)8 * 256 * 224 * 2);
  half_t* UH   = (half_t*)alloc((size_t)4 * 224 * 256 * 2);
  float*  SIM  = (float*)alloc((size_t)4 * 224 * 224 * 4);
  half_t* PH1  = (half_t*)alloc((size_t)4 * 224 * 224 * 2);
  half_t* PH2T = (half_t*)alloc((size_t)4 * 224 * 224 * 2);
  float*  ATT1 = (float*)alloc((size_t)4 * 224 * 256 * 4);
  float*  ATT2 = (float*)alloc((size_t)4 * 224 * 256 * 4);
  float*  DD   = (float*)alloc((size_t)2048 * 4);

  auto cvt = [&](const float* src, half_t* dst, size_t n) {
    k_cvt_f16<<<gsb(n), 256, 0, stream>>>(src, dst, n);
  };

  // ---- weight prep ----
  for (int s = 0; s < 4; ++s) {
    int pb = WN0 + 48 * s;
    half_t* sbase = WWN + s * STG_H;
    for (int j = 0; j < 7; ++j) {
      half_t* bb = sbase + j * BLK_H;
      cvt(f(pb + 6 * j + 0), bb, WG_H);                 // in_w  -> Wg
      cvt(f(pb + 6 * j + 4), bb + WG_H, W11_H);         // skip_w
      cvt(f(pb + 6 * j + 2), bb + WG_H + W11_H, W11_H); // res_w
    }
    cvt(f(pb + 42), sbase + 7 * BLK_H, W11_H);               // first
    cvt(f(pb + 44), sbase + 7 * BLK_H + W11_H, W11_H);       // last1
    cvt(f(pb + 46), sbase + 7 * BLK_H + 2 * W11_H, W11_H);   // last2
  }
  k_cvt_sinc<<<gsb(64 * 256), 256, 0, stream>>>(f(19), WSIN);
  k_lstm_pack<<<gsb(512 * 192), 256, 0, stream>>>(f(13), f(11), f(17), f(15), WL, BL);                   // fwd
  k_lstm_pack<<<gsb(512 * 192), 256, 0, stream>>>(f(12), f(10), f(16), f(14), WL + 512 * 192, BL + 512); // bwd
  k_cvt_T256<<<gsb(256 * 256), 256, 0, stream>>>(f(9), WCT);

  // ---- front end (WMMA GEMM, K=256) ----
  k_sinc_wmma<<<dim3((T1 + 63) / 64, 8), 256, 0, stream>>>(f(0), f(1), WSIN, f(18), PINH, T1);

  // ---- 4 WaveNet stages ----
  for (int s = 0; s < 4; ++s) {
    int T = Ts[s];
    size_t nT = (size_t)8 * 64 * T;
    half_t* sbase = WWN + s * STG_H;
    int pb = WN0 + 48 * s;
    float* Xf = A32; half_t* Xh = A16;
    float* Xof = B32; half_t* Xoh = B16;
    dim3 cg((T + 63) / 64, 8);

    k_zero_f32<<<gsb(nT), 256, 0, stream>>>(S32, nT);
    k_conv1x1<<<cg, 256, 0, stream>>>(PINH, sbase + 7 * BLK_H, f(pb + 43), Xf, Xh, T, 0);
    for (int j = 0; j < 7; ++j) {
      half_t* bb = sbase + j * BLK_H;
      k_wn_block<<<cg, 256, 0, stream>>>(Xh, Xf,
          bb, f(pb + 6 * j + 1),                       // Wg, bg
          bb + WG_H, f(pb + 6 * j + 5),                // Wsk, bsk
          bb + WG_H + W11_H, f(pb + 6 * j + 3),        // Wrs, brs
          S32, Xof, Xoh, T, 1 << j);
      float* tf = Xf; Xf = Xof; Xof = tf;
      half_t* th = Xh; Xh = Xoh; Xoh = th;
    }
    k_relu_cvt<<<gsb(nT), 256, 0, stream>>>(S32, Xoh, nT);   // relu(s) -> f16
    k_conv1x1<<<cg, 256, 0, stream>>>(Xoh, sbase + 7 * BLK_H + W11_H, f(pb + 45), nullptr, Xh, T, 1);
    k_conv1x1<<<cg, 256, 0, stream>>>(Xh, sbase + 7 * BLK_H + 2 * W11_H, f(pb + 47), Xof, nullptr, T, 0);
    k_pool<<<gsb((size_t)8 * 64 * Tp[s]), 256, 0, stream>>>(Xof, PIN32, PINH, T, Kp[s], Tp[s]);
  }

  // ---- BiLSTM ----
  k_lstm<<<2, 256, 0, stream>>>(PINH, WL, BL, ENC, 198);
  k_encpack<<<gsb((size_t)8 * 224 * 256), 256, 0, stream>>>(ENC, ENCH, ENCTH);

  // ---- co-attention ----
  const long long sE = 224 * 256, sS = 224 * 224;
  // u = y1 @ coatt_W   (f16 out)
  k_gemm<<<dim3(16, 14, 4), 32, 0, stream>>>(ENCH, 256, sE, WCT, 256, 0, nullptr, UH, 256, sE, 256);
  // sim = u @ y2^T
  k_gemm<<<dim3(14, 14, 4), 32, 0, stream>>>(UH, 256, sE, ENCH + 4 * sE, 256, sE, SIM, nullptr, 224, sS, 256);
  k_softmax2<<<dim3(224, 4), 256, 0, stream>>>(SIM, PH1, li(2), li(3));
  k_softmax1T<<<dim3(224, 4), 256, 0, stream>>>(SIM, PH2T, li(2), li(3));
  // a1 = P1 @ y2 ; a2 = P2^T @ y1
  k_gemm<<<dim3(16, 14, 4), 32, 0, stream>>>(PH1, 224, sS, ENCTH + 4 * sE, 224, sE, ATT1, nullptr, 256, sE, 224);
  k_gemm<<<dim3(16, 14, 4), 32, 0, stream>>>(PH2T, 224, sS, ENCTH, 224, sE, ATT2, nullptr, 256, sE, 224);

  // ---- distances, head, loss ----
  k_dsum<<<8, 256, 0, stream>>>(ATT1, ATT2, ENC, li(2), li(3), DD);
  k_head<<<1, 128, 0, stream>>>(DD, f(5), f(7), f(6), f(8), f(4), (float*)d_out);
}
